// DotProductAttention_75393855914174
// MI455X (gfx1250) — matrix operand order repaired and checked by reading, not compiled, not run on hardware
//
#include <hip/hip_runtime.h>

typedef __attribute__((ext_vector_type(16))) __bf16 v16bf;
typedef __attribute__((ext_vector_type(8)))  float  v8f;

#define LQ    4096
#define LK    4096
#define DHEAD 64
#define BQW   32     // query rows per wave (2 x 16-row M-tiles)
#define BQ    128    // queries per workgroup (4 waves x 32)
#define BK    128    // keys per chunk
#define NWAVE 4

union FragA { unsigned int u[8]; v16bf v; };
union FragC { float f[8];        v8f   v; };

// Max-reduce across the 16-lane group (lanes 0-15 / 16-31 independently).
// Butterfly with v_permlane16_b32: pure VALU, no LDS traffic, no dscnt waits.
__device__ __forceinline__ float rowmax16(float x) {
#if __has_builtin(__builtin_amdgcn_permlane16)
    union { float f; int i; } a, r;
    a.f = x;  // xor 1
    r.i = __builtin_amdgcn_permlane16(a.i, a.i, (int)0x67452301, (int)0xEFCDAB89, false, false);
    x = fmaxf(x, r.f);
    a.f = x;  // xor 2
    r.i = __builtin_amdgcn_permlane16(a.i, a.i, (int)0x54761032, (int)0xDCFE98BA, false, false);
    x = fmaxf(x, r.f);
    a.f = x;  // xor 4
    r.i = __builtin_amdgcn_permlane16(a.i, a.i, (int)0x32107654, (int)0xBA98FEDC, false, false);
    x = fmaxf(x, r.f);
    a.f = x;  // xor 8
    r.i = __builtin_amdgcn_permlane16(a.i, a.i, (int)0xFEDCBA98, (int)0x76543210, false, false);
    x = fmaxf(x, r.f);
    return x;
#else
    for (int sh = 1; sh < 16; sh <<= 1) x = fmaxf(x, __shfl_xor(x, sh, 32));
    return x;
#endif
}

__device__ __forceinline__ unsigned wave_any(bool p) {
#if __has_builtin(__builtin_amdgcn_ballot_w32)
    return __builtin_amdgcn_ballot_w32(p);
#else
    return (unsigned)__any(p);
#endif
}

__global__ __launch_bounds__(128)
void fa_wmma_bf16_kernel(const float* __restrict__ Q,
                         const float* __restrict__ K,
                         const float* __restrict__ V,
                         float* __restrict__ O)
{
    __shared__ __bf16 sK[BK * DHEAD];          // [key][dim]  bf16  (16 KB)
    __shared__ __bf16 sVt[DHEAD * BK];         // [dim][key]  bf16  (16 KB)
    __shared__ __bf16 sP[NWAVE][2][16 * BK];   // per-wave/M-tile P (32 KB)

    const int tid  = threadIdx.x;
    const int lane = tid & 31;
    const int wave = tid >> 5;
    const int half = lane >> 4;     // 0: lanes 0-15, 1: lanes 16-31
    const int ln   = lane & 15;

    const int b      = blockIdx.x / (LQ / BQ);
    const int qchunk = blockIdx.x % (LQ / BQ);
    const int qbase  = qchunk * BQ + wave * BQW;

    const float* Qb = Q + (size_t)b * LQ * DHEAD;
    const float* Kb = K + (size_t)b * LK * DHEAD;
    const float* Vb = V + (size_t)b * LK * DHEAD;
    float*       Ob = O + (size_t)b * LQ * DHEAD;

    // Fold 1/sqrt(d) and log2(e) into Q: scores land in the exp2 domain.
    // (zero-ness of a score is invariant under this scaling -> mask still valid)
    const float qscale = 0.125f * 1.44269504088896340736f;

    // ---- Q tiles (2 x 16 x 64 f32) -> bf16 A-fragments ----
    FragA qa[2][2];
#pragma unroll
    for (int m = 0; m < 2; ++m) {
        const unsigned qoff = (unsigned)(qbase + m * 16 + ln) * DHEAD;
#pragma unroll
        for (int f = 0; f < 2; ++f) {
#pragma unroll
            for (int v = 0; v < 8; ++v) {
                int kb  = ((v < 4) ? 0 : 16) + half * 8 + (v & 3) * 2;
                float2 qq = *reinterpret_cast<const float2*>(&Qb[qoff + f * 32 + kb]);
                union { __bf16 h[2]; unsigned int u; } pk;
                pk.h[0] = (__bf16)(qq.x * qscale);
                pk.h[1] = (__bf16)(qq.y * qscale);
                qa[m][f].u[v] = pk.u;
            }
        }
    }

    // all-ones bf16 B-fragment for WMMA row-sum (P . 1 -> rowsum in every column)
    FragA ones;
#pragma unroll
    for (int v = 0; v < 8; ++v) ones.u[v] = 0x3F803F80u;

    FragC oacc[2][4];
#pragma unroll
    for (int m = 0; m < 2; ++m)
#pragma unroll
        for (int t = 0; t < 4; ++t)
#pragma unroll
            for (int i = 0; i < 8; ++i) oacc[m][t].f[i] = 0.f;

    float mrow[2][8], lrow[2][8];
#pragma unroll
    for (int m = 0; m < 2; ++m)
#pragma unroll
        for (int r = 0; r < 8; ++r) { mrow[m][r] = -INFINITY; lrow[m][r] = 0.f; }

    // loop-invariant 32-bit staging offsets (uniform base + vector offset + imm)
    const unsigned kto = ((unsigned)(tid >> 4) << 6) | ((unsigned)(tid & 15) << 2);
    const unsigned vto = ((unsigned)(tid >> 4) << 7) | ((unsigned)(tid & 15) << 2);
    const unsigned svo = ((unsigned)(tid & 15) << 9) | ((unsigned)(tid >> 4) << 1);

    for (int kc = 0; kc < LK / BK; ++kc) {
        const float* Kp = Kb + kc * (BK * DHEAD);
        const float* Vp = Vb + kc * (BK * DHEAD);
        __syncthreads();   // previous iteration's LDS readers done

        // ---- stage K (bf16 row-major): 128x64, packed b64 LDS stores ----
#pragma unroll
        for (int i = 0; i < 16; ++i) {
            float4 kk = *reinterpret_cast<const float4*>(Kp + kto + i * 512);
            union { __bf16 h[4]; uint2 u2; } pk4;
            pk4.h[0] = (__bf16)kk.x; pk4.h[1] = (__bf16)kk.y;
            pk4.h[2] = (__bf16)kk.z; pk4.h[3] = (__bf16)kk.w;
            *reinterpret_cast<uint2*>(&sK[kto + i * 512]) = pk4.u2;
        }
        // ---- stage V transposed (bf16 [dim][key]): key-pairs -> packed b32 ----
#pragma unroll
        for (int i = 0; i < 8; ++i) {
            float4 v0 = *reinterpret_cast<const float4*>(Vp + vto + i * 1024);
            float4 v1 = *reinterpret_cast<const float4*>(Vp + vto + i * 1024 + DHEAD);
            union { __bf16 h[2]; unsigned int u; } a0, a1, a2, a3;
            a0.h[0] = (__bf16)v0.x; a0.h[1] = (__bf16)v1.x;
            a1.h[0] = (__bf16)v0.y; a1.h[1] = (__bf16)v1.y;
            a2.h[0] = (__bf16)v0.z; a2.h[1] = (__bf16)v1.z;
            a3.h[0] = (__bf16)v0.w; a3.h[1] = (__bf16)v1.w;
            *reinterpret_cast<unsigned int*>(&sVt[svo + i * 16])       = a0.u;
            *reinterpret_cast<unsigned int*>(&sVt[svo + i * 16 + BK])      = a1.u;
            *reinterpret_cast<unsigned int*>(&sVt[svo + i * 16 + 2 * BK]) = a2.u;
            *reinterpret_cast<unsigned int*>(&sVt[svo + i * 16 + 3 * BK]) = a3.u;
        }
        __syncthreads();

        // ==== per M-tile: S, softmax, P store (one sacc set live at a time) ====
#pragma unroll
        for (int m = 0; m < 2; ++m) {
            // ---- S = Qs . K^T : 8 interleaved key tiles ----
            // tile t covers keys 32*(t>>1) + 2*n + (t&1), so a lane's
            // (sacc[2j],sacc[2j+1]) pairs hold adjacent keys.
            FragC sacc[8];
#pragma unroll
            for (int t = 0; t < 8; ++t) {
                const int key = 32 * (t >> 1) + 2 * ln + (t & 1);
                v8f c = {};
#pragma unroll
                for (int h = 0; h < 2; ++h) {
                    FragA bk;
#pragma unroll
                    for (int v = 0; v < 8; ++v) {
                        int kdim = h * 32 + ((v < 4) ? 0 : 16) + half * 8 + (v & 3) * 2;
                        bk.u[v] = *reinterpret_cast<const unsigned int*>(
                            &sK[key * DHEAD + kdim]);
                    }
                    c = __builtin_amdgcn_wmma_f32_16x16x32_bf16(
                            false, qa[m][h].v, false, bk.v, (short)0, c, false, false);
                }
                sacc[t].v = c;
            }

            // ---- zero-mask, online max (VALU butterfly), p = exp2(s - m) ----
            float newm[8];
            bool need = false;
#pragma unroll
            for (int r = 0; r < 8; ++r) {
                float s[8];
#pragma unroll
                for (int t = 0; t < 8; ++t) {
                    float sv = sacc[t].f[r];
                    if (sv == 0.f) sv = -1e30f;   // reference: scores==0 -> -1e30
                    s[t] = sv;
                }
                float mx01 = fmaxf(s[0], s[1]), mx23 = fmaxf(s[2], s[3]);
                float mx45 = fmaxf(s[4], s[5]), mx67 = fmaxf(s[6], s[7]);
                float mx = rowmax16(fmaxf(fmaxf(mx01, mx23), fmaxf(mx45, mx67)));
                float mnew = fmaxf(mrow[m][r], mx);
                newm[r] = mnew;
                need = need || (mnew > mrow[m][r]);
#pragma unroll
                for (int t = 0; t < 8; ++t)
                    sacc[t].f[r] = __builtin_amdgcn_exp2f(s[t] - mnew);
            }
            // rescale only when some row's running max actually moved (corr != 1)
            if (wave_any(need)) {
#pragma unroll
                for (int r = 0; r < 8; ++r) {
                    float corr = __builtin_amdgcn_exp2f(mrow[m][r] - newm[r]);
                    lrow[m][r] *= corr;
                    oacc[m][0].f[r] *= corr;
                    oacc[m][1].f[r] *= corr;
                    oacc[m][2].f[r] *= corr;
                    oacc[m][3].f[r] *= corr;
                    mrow[m][r] = newm[r];
                }
            }

            // ---- P -> per-wave LDS (packed key pairs; natural key order) ----
            // sP is private to this wave; a wave's LDS ops execute in order,
            // so no workgroup barrier is needed between these stores and loads.
#pragma unroll
            for (int r = 0; r < 8; ++r) {
                int row = r + half * 8;
#pragma unroll
                for (int j = 0; j < 4; ++j) {
                    union { __bf16 h[2]; unsigned int u; } p;
                    p.h[0] = (__bf16)sacc[2 * j].f[r];
                    p.h[1] = (__bf16)sacc[2 * j + 1].f[r];
                    *reinterpret_cast<unsigned int*>(
                        &sP[wave][m][row * BK + 32 * j + 2 * ln]) = p.u;
                }
            }
        }

        // ---- P A-fragments for both M-tiles ----
        FragA pa[2][4];
#pragma unroll
        for (int m = 0; m < 2; ++m)
#pragma unroll
            for (int kf = 0; kf < 4; ++kf)
#pragma unroll
                for (int v = 0; v < 8; ++v) {
                    int kb = kf * 32 + ((v < 4) ? 0 : 16) + half * 8 + (v & 3) * 2;
                    pa[m][kf].u[v] = *reinterpret_cast<const unsigned int*>(
                        &sP[wave][m][ln * BK + kb]);
                }

        // ---- row sums via WMMA: psum = P . ones (rowsum in every column) ----
#pragma unroll
        for (int m = 0; m < 2; ++m) {
            v8f ps = {};
#pragma unroll
            for (int kf = 0; kf < 4; ++kf)
                ps = __builtin_amdgcn_wmma_f32_16x16x32_bf16(
                        false, pa[m][kf].v, false, ones.v, (short)0, ps, false, false);
            FragC psum; psum.v = ps;
#pragma unroll
            for (int r = 0; r < 8; ++r)
                lrow[m][r] += psum.f[r];
        }

        // ---- O += P . V : each bv fragment feeds both M-tiles ----
#pragma unroll
        for (int t = 0; t < 4; ++t) {
#pragma unroll
            for (int kf = 0; kf < 4; ++kf) {
                FragA bv;
#pragma unroll
                for (int v = 0; v < 8; ++v) {
                    int key = kf * 32 + ((v < 4) ? 0 : 16) + half * 8 + (v & 3) * 2;
                    int dim = t * 16 + ln;
                    bv.u[v] = *reinterpret_cast<const unsigned int*>(
                        &sVt[dim * BK + key]);
                }
                oacc[0][t].v = __builtin_amdgcn_wmma_f32_16x16x32_bf16(
                    false, pa[0][kf].v, false, bv.v, (short)0, oacc[0][t].v, false, false);
                oacc[1][t].v = __builtin_amdgcn_wmma_f32_16x16x32_bf16(
                    false, pa[1][kf].v, false, bv.v, (short)0, oacc[1][t].v, false, false);
            }
        }
    }

    // ---- finalize: O / l, store fp32 ----
#pragma unroll
    for (int m = 0; m < 2; ++m)
#pragma unroll
        for (int r = 0; r < 8; ++r) {
            float inv = 1.f / lrow[m][r];
            unsigned row = (unsigned)(qbase + m * 16 + r + half * 8);
#pragma unroll
            for (int t = 0; t < 4; ++t) {
                Ob[row * DHEAD + t * 16 + ln] = oacc[m][t].f[r] * inv;
            }
        }
}

extern "C" void kernel_launch(void* const* d_in, const int* in_sizes, int n_in,
                              void* d_out, int out_size, void* d_ws, size_t ws_size,
                              hipStream_t stream)
{
    (void)in_sizes; (void)n_in; (void)out_size; (void)d_ws; (void)ws_size;
    const float* Q = (const float*)d_in[0];
    const float* K = (const float*)d_in[1];
    const float* V = (const float*)d_in[2];
    float*       O = (float*)d_out;

    const int B = 16;
    dim3 grid(B * (LQ / BQ));   // 16 * 32 = 512 workgroups
    dim3 block(128);            // 4 wave32
    fa_wmma_bf16_kernel<<<grid, block, 0, stream>>>(Q, K, V, O);
}
